// MLPPredictor_89000312308383
// MI455X (gfx1250) — matrix operand order repaired and checked by reading, not compiled, not run on hardware
//
#include <hip/hip_runtime.h>

typedef _Float16 v16h __attribute__((ext_vector_type(16)));
typedef _Float16 v8h  __attribute__((ext_vector_type(8)));
typedef float    v8f  __attribute__((ext_vector_type(8)));

#define HF      128          // feature width
#define PAD     136          // padded LDS row length in halves (+8 -> conflict-free)
#define TILE_E  64           // edges per workgroup tile
#define BLOCK   128          // 4 waves of 32

__global__ __launch_bounds__(BLOCK)
void edge_mlp_wmma(const float* __restrict__ h,
                   const int*   __restrict__ src,
                   const int*   __restrict__ dst,
                   const float* __restrict__ W1,
                   const float* __restrict__ b1,
                   const float* __restrict__ W2,
                   const float* __restrict__ b2,
                   float*       __restrict__ out,
                   int nEdges, int nTiles)
{
    __shared__ __align__(16) _Float16 w1t[HF * PAD];      // W1 transposed: [n][k] f16
    __shared__ __align__(16) _Float16 eA [TILE_E * PAD];  // edge features: [m][k] f16
    __shared__ float lb1[HF];
    __shared__ float lw2[HF];

    const int t    = threadIdx.x;
    const int lane = t & 31;
    const int wave = t >> 5;
    const int hsel = lane >> 4;    // 0: lanes 0-15, 1: lanes 16-31
    const int l16  = lane & 15;

    // ---- one-time per block: stage W1^T (f16), b1, W2 into LDS ----
    for (int idx = t; idx < HF * HF; idx += BLOCK) {
        int k = idx >> 7;          // W1 row (input feature)
        int n = idx & (HF - 1);    // W1 col (output feature)
        w1t[n * PAD + k] = (_Float16)W1[idx];
    }
    if (t < HF) { lb1[t] = b1[t]; lw2[t] = W2[t]; }
    __syncthreads();

    const float b2v = b2[0];

    for (int tile = blockIdx.x; tile < nTiles; tile += (int)gridDim.x) {
        const int e0 = tile * TILE_E;

        // ---- gather h[src]*h[dst] -> eA (f16) ----
        // Each wave gathers its own 16 edges; edge index is wave-uniform so
        // src[e]/dst[e] scalarize (s_load) and the 32 lanes load one full
        // 512-B node row per instruction (lane = float4 slice). Fully
        // coalesced L2 gathers on the bandwidth-bound stage.
        {
            const int mb = wave * 16;
            #pragma unroll 4
            for (int i = 0; i < 16; ++i) {
                const int m = mb + i;
                const int e = e0 + m;              // uniform across wave
                if (e < nEdges) {
                    const int s = src[e];
                    const int d = dst[e];
                    float4 a = ((const float4*)(h + (size_t)s * HF))[lane];
                    float4 b = ((const float4*)(h + (size_t)d * HF))[lane];
                    union { _Float16 hv[4]; uint2 u; } p;
                    p.hv[0] = (_Float16)(a.x * b.x);
                    p.hv[1] = (_Float16)(a.y * b.y);
                    p.hv[2] = (_Float16)(a.z * b.z);
                    p.hv[3] = (_Float16)(a.w * b.w);
                    *(uint2*)(&eA[m * PAD + lane * 4]) = p.u;
                } else {
                    const uint2 z = {0u, 0u};
                    *(uint2*)(&eA[m * PAD + lane * 4]) = z;
                }
            }
        }
        __syncthreads();

        // ---- layer 1: each wave does a 16x128 stripe via 32 WMMAs (4 K-steps x 8 N-tiles) ----
        v8f acc[8] = {};
        const int row = wave * 16 + l16;      // A-matrix row (edge slot) for this lane

        #pragma unroll
        for (int kk = 0; kk < 4; ++kk) {
            const int k0 = kk * 32;
            // A fragment (16x32 f16): per ISA table, lane holds K = 8*hsel+0..7 then +16
            v8h a0 = *(const v8h*)(&eA[row * PAD + k0 + 8 * hsel]);
            v8h a1 = *(const v8h*)(&eA[row * PAD + k0 + 8 * hsel + 16]);
            v16h afrag = __builtin_shufflevector(a0, a1,
                0,1,2,3,4,5,6,7,8,9,10,11,12,13,14,15);
            #pragma unroll
            for (int nt = 0; nt < 8; ++nt) {
                const int n = nt * 16 + l16;
                // B fragment (32x16 f16): lane holds K = 8*hsel+0..7 then +16 (matches A)
                v8h b0  = *(const v8h*)(&w1t[n * PAD + k0 + 8 * hsel]);
                v8h b1f = *(const v8h*)(&w1t[n * PAD + k0 + 8 * hsel + 16]);
                v16h bfrag = __builtin_shufflevector(b0, b1f,
                    0,1,2,3,4,5,6,7,8,9,10,11,12,13,14,15);
                acc[nt] = __builtin_amdgcn_wmma_f32_16x16x32_f16(
                    false, afrag, false, bfrag, (short)0, acc[nt], false, false);
            }
        }

        // ---- layer 2: bias + ReLU + dot with W2 on accumulator layout ----
        // C layout: VGPR v -> (M = v + 8*hsel, N = l16) within the 16x16 tile.
        float s[8];
        #pragma unroll
        for (int v = 0; v < 8; ++v) s[v] = 0.f;
        #pragma unroll
        for (int nt = 0; nt < 8; ++nt) {
            const int n    = nt * 16 + l16;
            const float wv = lw2[n];
            const float bv = lb1[n];
            #pragma unroll
            for (int v = 0; v < 8; ++v) {
                float a = acc[nt][v] + bv;
                a = a > 0.f ? a : 0.f;
                s[v] += a * wv;
            }
        }
        // reduce over N across each 16-lane half (xor masks 1..8 stay within halves)
        #pragma unroll
        for (int off = 1; off < 16; off <<= 1) {
            #pragma unroll
            for (int v = 0; v < 8; ++v)
                s[v] += __shfl_xor(s[v], off, 32);
        }
        if (l16 == 0) {
            const int mbase = wave * 16 + hsel * 8;
            #pragma unroll
            for (int v = 0; v < 8; ++v) {
                const int e = e0 + mbase + v;
                if (e < nEdges) out[e] = s[v] + b2v;
            }
        }
        __syncthreads();   // eA reused next tile
    }
}

extern "C" void kernel_launch(void* const* d_in, const int* in_sizes, int n_in,
                              void* d_out, int out_size, void* d_ws, size_t ws_size,
                              hipStream_t stream) {
    const float* h   = (const float*)d_in[0];
    const int*   src = (const int*)  d_in[1];
    const int*   dst = (const int*)  d_in[2];
    const float* W1  = (const float*)d_in[3];
    const float* b1  = (const float*)d_in[4];
    const float* W2  = (const float*)d_in[5];
    const float* b2  = (const float*)d_in[6];
    float* out = (float*)d_out;

    const int nEdges = in_sizes[1];
    const int nTiles = (nEdges + TILE_E - 1) / TILE_E;
    const int blocks = nTiles < 2560 ? nTiles : 2560;   // persistent blocks amortize W1 staging

    hipLaunchKernelGGL(edge_mlp_wmma, dim3(blocks), dim3(BLOCK), 0, stream,
                       h, src, dst, W1, b1, W2, b2, out, nEdges, nTiles);
}
